// _AA_CostVolume_63668595196248
// MI455X (gfx1250) — matrix operand order repaired and checked by reading, not compiled, not run on hardware
//
#include <hip/hip_runtime.h>

typedef __bf16 bf16;
typedef __attribute__((ext_vector_type(4)))  __bf16 v4bf;
typedef __attribute__((ext_vector_type(8)))  __bf16 v8bf;
typedef __attribute__((ext_vector_type(16))) __bf16 v16bf;
typedef __attribute__((ext_vector_type(8)))  float  v8f;

#define CH    256
#define HDIM  180
#define WDIM  320
#define NDISP 64
#define HW    (HDIM*WDIM)          /* 57600 : channel stride */
#define CHW   ((size_t)CH*HW)      /* per-batch input stride */

// One workgroup (4 wave32) per (b, h, 64-wide x strip).
// cost[b,i,h,x] = (1/256) * sum_c L[b,c,h,x] * R[b,c,h,x-i], 0 <= i < 64, zero for x<i.
__global__ __launch_bounds__(128)
void _AA_CostVolume_wmma_kernel(const float* __restrict__ Lf,
                                const float* __restrict__ Rf,
                                float* __restrict__ out)
{
    // Channel-contiguous rows; +8 bf16 pad -> 528B pitch (conflict-free b128 frag loads)
    __shared__ bf16  sL[64][264];    // left  columns x  in [X0, X0+64)
    __shared__ bf16  sR[128][264];   // right columns x' in [X0-64, X0+64)
    __shared__ float sOut[64][68];   // [disparity i][x - X0]

    const int s  = blockIdx.x;       // x strip: 0..4
    const int h  = blockIdx.y;
    const int b  = blockIdx.z;
    const int X0 = 64 * s;

    const float* Lb = Lf + (size_t)b * CHW + (size_t)h * WDIM;
    const float* Rb = Rf + (size_t)b * CHW + (size_t)h * WDIM;

    const int lane = threadIdx.x & 31;
    const int wv   = threadIdx.x >> 5;   // wave id 0..3

    // ---- Stage + transpose + f32->bf16 in 4x4 register micro-tiles ----------
    // Per lane-iter: 4 coalesced float4 global loads (x-contiguous),
    // 4 ds_store_b64 (4 bf16 along c). Wave covers 8 x-blocks x 4 c-blocks.
    {
        const int xbo = lane & 7;        // x-block offset within group
        const int cbo = lane >> 3;       // c-block offset within group

        // Left: 64 x  -> 16 x-blocks (2 groups), 256 c -> 64 c-blocks (16 groups)
        for (int u = wv; u < 2 * 16; u += 4) {
            int xg = u & 1, cg = u >> 1;
            int x4 = 4 * (8 * xg + xbo);             // 0..60
            int c0 = 4 * (4 * cg + cbo);             // 0..252
            const float* p = &Lb[(size_t)c0 * HW + X0 + x4];
            float4 f0 = *(const float4*)(p + 0 * HW);
            float4 f1 = *(const float4*)(p + 1 * HW);
            float4 f2 = *(const float4*)(p + 2 * HW);
            float4 f3 = *(const float4*)(p + 3 * HW);
            v4bf r0 = { (bf16)f0.x, (bf16)f1.x, (bf16)f2.x, (bf16)f3.x };
            v4bf r1 = { (bf16)f0.y, (bf16)f1.y, (bf16)f2.y, (bf16)f3.y };
            v4bf r2 = { (bf16)f0.z, (bf16)f1.z, (bf16)f2.z, (bf16)f3.z };
            v4bf r3 = { (bf16)f0.w, (bf16)f1.w, (bf16)f2.w, (bf16)f3.w };
            *(v4bf*)&sL[x4 + 0][c0] = r0;
            *(v4bf*)&sL[x4 + 1][c0] = r1;
            *(v4bf*)&sL[x4 + 2][c0] = r2;
            *(v4bf*)&sL[x4 + 3][c0] = r3;
        }
        // Right: 128 x' -> 32 x-blocks (4 groups), 256 c -> 64 c-blocks (16 groups)
        for (int u = wv; u < 4 * 16; u += 4) {
            int xg = u & 3, cg = u >> 2;
            int x4 = 4 * (8 * xg + xbo);             // 0..124
            int c0 = 4 * (4 * cg + cbo);
            int gx = X0 - 64 + x4;                   // multiple of 4; neg region is whole tiles
            float4 f0, f1, f2, f3;
            if (gx >= 0) {
                const float* p = &Rb[(size_t)c0 * HW + gx];
                f0 = *(const float4*)(p + 0 * HW);
                f1 = *(const float4*)(p + 1 * HW);
                f2 = *(const float4*)(p + 2 * HW);
                f3 = *(const float4*)(p + 3 * HW);
            } else {
                f0 = f1 = f2 = f3 = make_float4(0.f, 0.f, 0.f, 0.f);
            }
            v4bf r0 = { (bf16)f0.x, (bf16)f1.x, (bf16)f2.x, (bf16)f3.x };
            v4bf r1 = { (bf16)f0.y, (bf16)f1.y, (bf16)f2.y, (bf16)f3.y };
            v4bf r2 = { (bf16)f0.z, (bf16)f1.z, (bf16)f2.z, (bf16)f3.z };
            v4bf r3 = { (bf16)f0.w, (bf16)f1.w, (bf16)f2.w, (bf16)f3.w };
            *(v4bf*)&sR[x4 + 0][c0] = r0;
            *(v4bf*)&sR[x4 + 1][c0] = r1;
            *(v4bf*)&sR[x4 + 2][c0] = r2;
            *(v4bf*)&sR[x4 + 3][c0] = r3;
        }
    }
    __syncthreads();

    // ---- Banded GEMM via WMMA: D_j[m,n] = sum_c R[c, x'=Xt-16j+m] * L[c, x=Xt+n] ----
    const int nm = lane & 15;            // A: row M / B,C,D: col N
    const int hi = lane >> 4;            // K-half selector per ISA VGPR layouts

    auto loadB = [&](int k0) -> v16bf {  // left tile: lanes0-15 K=k0..+15, lanes16-31 K=k0+16..+31
        const bf16* bp = &sL[16 * wv + nm][k0 + 16 * hi];
        v8bf b0 = *(const v8bf*)bp;
        v8bf b1 = *(const v8bf*)(bp + 8);
        return __builtin_shufflevector(b0, b1,
            0,1,2,3,4,5,6,7,8,9,10,11,12,13,14,15);
    };
    auto loadA = [&](int j, int k0) -> v16bf {  // right tile j: lanes0-15 K=k0..+15, lanes16-31 K=k0+16..+31
        const bf16* ap = &sR[64 + 16 * (wv - j) + nm][k0 + 16 * hi];
        v8bf a0 = *(const v8bf*)ap;
        v8bf a1 = *(const v8bf*)(ap + 8);
        return __builtin_shufflevector(a0, a1,
            0,1,2,3,4,5,6,7,8,9,10,11,12,13,14,15);
    };

    v8f acc[5] = {};
    v16bf bcur = loadB(0);
    v16bf acur[5];
    #pragma unroll
    for (int j = 0; j < 5; ++j) acur[j] = loadA(j, 0);

    #pragma unroll
    for (int t = 0; t < 8; ++t) {
        v16bf bnxt;
        v16bf anxt[5];
        if (t < 7) {                      // prefetch next k-step's fragments first
            bnxt = loadB(32 * (t + 1));
            #pragma unroll
            for (int j = 0; j < 5; ++j) anxt[j] = loadA(j, 32 * (t + 1));
        }
        #pragma unroll
        for (int j = 0; j < 5; ++j)       // 5 WMMAs back-to-back, no per-op dscnt-0 wait
            acc[j] = __builtin_amdgcn_wmma_f32_16x16x32_bf16(
                false, acur[j], false, bcur, (short)0, acc[j], false, false);
        if (t < 7) {
            bcur = bnxt;
            #pragma unroll
            for (int j = 0; j < 5; ++j) acur[j] = anxt[j];
        }
    }

    // ---- Scatter diagonal tiles into (i, x) layout in LDS (each (i,x) hit once) ----
    #pragma unroll
    for (int j = 0; j < 5; ++j) {
        #pragma unroll
        for (int r = 0; r < 8; ++r) {
            int m = r + 8 * hi;                  // C/D layout: VGPR r -> M=r (+8 for hi lanes)
            int i = 16 * j + nm - m;             // disparity
            if (i >= 0 && i < NDISP)
                sOut[i][16 * wv + nm] = acc[j][r] * (1.0f / 256.0f);
        }
    }
    __syncthreads();

    // ---- Coalesced float4 store of the 64x64 (i, x) tile ----
    float* Ob = out + (size_t)b * NDISP * HW + (size_t)h * WDIM + X0;
    for (int idx = threadIdx.x; idx < 64 * 16; idx += 128) {
        int i = idx >> 4, seg = idx & 15;
        float4 v = *(const float4*)&sOut[i][4 * seg];
        *(float4*)&Ob[(size_t)i * HW + 4 * seg] = v;
    }
}

extern "C" void kernel_launch(void* const* d_in, const int* in_sizes, int n_in,
                              void* d_out, int out_size, void* d_ws, size_t ws_size,
                              hipStream_t stream) {
    const float* L = (const float*)d_in[0];
    const float* R = (const float*)d_in[1];
    float* out = (float*)d_out;
    dim3 grid(5, HDIM, 4);           // x-strips, h, b
    _AA_CostVolume_wmma_kernel<<<grid, 128, 0, stream>>>(L, R, out);
}
